// Model_20839181320549
// MI455X (gfx1250) — matrix operand order repaired and checked by reading, not compiled, not run on hardware
//
#include <hip/hip_runtime.h>
#include <hip/hip_bf16.h>
#include <math.h>

#define Bn 8
#define Tn 96
#define Nn 20000
#define Pn 12
#define Ln 3
#define Dn 32
#define Cn 128
#define Mn 64
#define NT ((Nn + 127) / 128)        /* 157 row-tiles per batch */
#define INV_SQRT_TAU 2.0f            /* 1/sqrt(0.25) */
#define S4 0.4204482076268573f       /* 32^-0.25 */
#define RATIO 0.125f                 /* 1/sqrt(64) */
#define EPSF 1e-6f

typedef __attribute__((ext_vector_type(16))) unsigned short v16u;
typedef __attribute__((ext_vector_type(16))) __bf16        v16bf;
typedef __attribute__((ext_vector_type(8)))  float         v8f;

__device__ inline unsigned short f2bf(float f){
  unsigned int u = __float_as_uint(f);
  u += 0x7FFFu + ((u >> 16) & 1u);
  return (unsigned short)(u >> 16);
}
__device__ inline float bf2f(unsigned short h){ return __uint_as_float(((unsigned int)h) << 16); }

__device__ inline unsigned int fenc(float f){
  unsigned int u = __float_as_uint(f);
  return (u & 0x80000000u) ? ~u : (u | 0x80000000u);
}
__device__ inline float fdec(unsigned int e){
  unsigned int u = (e & 0x80000000u) ? (e & 0x7FFFFFFFu) : ~e;
  return __uint_as_float(u);
}

__device__ inline float wsum(float v){
  #pragma unroll
  for (int m = 16; m >= 1; m >>= 1) v += __shfl_xor(v, m, 32);
  return v;
}
__device__ inline float wmaxr(float v){
  #pragma unroll
  for (int m = 16; m >= 1; m >>= 1) v = fmaxf(v, __shfl_xor(v, m, 32));
  return v;
}
__device__ inline float hsum16(float v){      // reduce within 16-lane half of a wave32
  #pragma unroll
  for (int m = 8; m >= 1; m >>= 1) v += __shfl_xor(v, m, 32);
  return v;
}
__device__ inline v8f splat8(float x){ v8f v = {x,x,x,x,x,x,x,x}; return v; }

// A fragment (16x32 bf16), storage row-major [m][k]:
//   lane L: m = rowbase + L%16; halves 0..7 -> k = kbase + 8*(L>=16) + j
//           halves 8..15 -> k = kbase + 16 + 8*(L>=16) + (j-8)
__device__ inline v16u ldA(const unsigned short* base, int ld, int rowbase, int kbase){
  int lane = threadIdx.x & 31; int hi = lane >> 4;
  const unsigned short* p = base + (size_t)(rowbase + (lane & 15)) * ld + kbase + hi * 8;
  union { v16u v; uint4 q[2]; } u;
  u.q[0] = *(const uint4*)p;
  u.q[1] = *(const uint4*)(p + 16);
  return u.v;
}
// B fragment (32x16 bf16), storage as [n][k]:
//   lane L: n = colbase + L%16; halves 0..7 -> k = kbase + 8*(L>=16) + j,
//           halves 8..15 -> k = kbase + 16 + 8*(L>=16) + (j-8)
__device__ inline v16u ldB(const unsigned short* base, int ld, int colbase, int kbase){
  int lane = threadIdx.x & 31; int hi = lane >> 4;
  const unsigned short* p = base + (size_t)(colbase + (lane & 15)) * ld + kbase + hi * 8;
  union { v16u v; uint4 q[2]; } u;
  u.q[0] = *(const uint4*)p;
  u.q[1] = *(const uint4*)(p + 16);
  return u.v;
}
__device__ inline v16u ldA_guard(const unsigned short* base, int ld, int rowbase, int kbase, int nrows){
  int lane = threadIdx.x & 31; int hi = lane >> 4;
  int r = rowbase + (lane & 15);
  union { v16u v; uint4 q[2]; } u;
  if (r < nrows){
    const unsigned short* p = base + (size_t)r * ld + kbase + hi * 8;
    u.q[0] = *(const uint4*)p;
    u.q[1] = *(const uint4*)(p + 16);
  } else {
    uint4 z; z.x = 0u; z.y = 0u; z.z = 0u; z.w = 0u;
    u.q[0] = z; u.q[1] = z;
  }
  return u.v;
}

__device__ inline v8f wmma_bf16(v16u a, v16u b, v8f c){
  return __builtin_amdgcn_wmma_f32_16x16x32_bf16(
      false, __builtin_bit_cast(v16bf, a),
      false, __builtin_bit_cast(v16bf, b),
      (short)0, c, false, false);
}

// Async copy of one 16-byte packet per lane: global -> LDS (ASYNCcnt-tracked).
__device__ inline void async_g2l_b128(unsigned lds_byte_off, const void* gaddr){
  asm volatile("global_load_async_to_lds_b128 %0, %1, off"
               :: "v"(lds_byte_off), "v"((unsigned long long)(size_t)gaddr)
               : "memory");
}
__device__ inline void wait_async0(){
  asm volatile("s_wait_asynccnt 0x0" ::: "memory");
}

// ---------------------------------------------------------------------------
// Per-batch prep: indices, time/week rows, mark-channel bias, nv biases, Dbq/Dbk
__global__ __launch_bounds__(64) void prep_batch_kernel(
    const float* __restrict__ xmark, const float* __restrict__ time_tab,
    const float* __restrict__ week_tab, const float* __restrict__ input_w,
    const float* __restrict__ input_b, const float* __restrict__ w1w,
    const float* __restrict__ w1b, const float* __restrict__ w2w,
    const float* __restrict__ w2b, const float* __restrict__ proj,
    float* te, float* we, float* markc, float* bias1, float* bias2,
    float* Dbq, float* Dbk)
{
  __shared__ float teS[Dn], weS[Dn], b1S[Dn], b2S[Dn];
  __shared__ int idxS[2];
  int b = blockIdx.x, t = threadIdx.x;
  if (t == 0){
    float m0 = xmark[((b * Tn) + (Tn - 1)) * 2 + 0];
    float m1 = xmark[((b * Tn) + (Tn - 1)) * 2 + 1];
    int tod = (int)(m0 * (float)Tn); tod = max(0, min(Tn - 1, tod));
    int dow = (int)(m1 * 7.0f);      dow = max(0, min(6, dow));
    idxS[0] = tod; idxS[1] = dow;
  }
  __syncthreads();
  if (t < Dn){
    float tv = time_tab[idxS[0] * Dn + t];
    float wv = week_tab[idxS[1] * Dn + t];
    teS[t] = tv; weS[t] = wv;
    te[b * Dn + t] = tv; we[b * Dn + t] = wv;
    float a = input_b[t];
    for (int tt = 0; tt < Tn; ++tt){
      float x0 = xmark[((b * Tn) + tt) * 2 + 0];
      float x1 = xmark[((b * Tn) + tt) * 2 + 1];
      a += x0 * input_w[t * (3 * Tn) + Tn + tt];
      a += x1 * input_w[t * (3 * Tn) + 2 * Tn + tt];
    }
    markc[b * Dn + t] = a;
  }
  __syncthreads();
  if (t < Dn){
    float v1 = w1b[t], v2 = w2b[t];
    for (int j = 0; j < Dn; ++j){
      v1 += teS[j] * w1w[t * (3 * Dn) + Dn + j] + weS[j] * w1w[t * (3 * Dn) + 2 * Dn + j];
      v2 += teS[j] * w2w[t * (3 * Dn) + Dn + j] + weS[j] * w2w[t * (3 * Dn) + 2 * Dn + j];
    }
    v1 *= INV_SQRT_TAU; v2 *= INV_SQRT_TAU;
    b1S[t] = v1; b2S[t] = v2;
    bias1[b * Dn + t] = v1; bias2[b * Dn + t] = v2;
  }
  __syncthreads();
  if (t < Mn){
    float dq = 0.f, dk = 0.f;
    #pragma unroll
    for (int c = 0; c < Dn; ++c){
      float p = proj[t * Dn + c];
      dq += b1S[c] * p; dk += b2S[c] * p;
    }
    Dbq[b * Mn + t] = dq * S4;
    Dbk[b * Mn + t] = dk * S4;
  }
}

// ---------------------------------------------------------------------------
// Per-node prep: NE1/NE2 (node part of nv1/nv2), Dqn/Dkn, per-m global max of Dkn
__global__ __launch_bounds__(64) void prep_node_kernel(
    const float* __restrict__ node_emb, const float* __restrict__ w1w,
    const float* __restrict__ w2w, const float* __restrict__ proj,
    float* NE1, float* NE2, float* Dqn, float* Dkn, unsigned int* maxkU)
{
  __shared__ float w1s[Dn][Dn], w2s[Dn][Dn], ps[Mn][Dn];
  __shared__ unsigned int bmax[Mn];
  int tid = threadIdx.x;
  for (int i = tid; i < Dn * Dn; i += 64){
    int c = i >> 5, g = i & 31;
    w1s[c][g] = w1w[c * (3 * Dn) + g];
    w2s[c][g] = w2w[c * (3 * Dn) + g];
  }
  for (int i = tid; i < Mn * Dn; i += 64) ps[i >> 5][i & 31] = proj[i];
  if (tid < Mn) bmax[tid] = 0u;
  __syncthreads();
  int n = blockIdx.x * 64 + tid;
  if (n < Nn){
    float ne[Dn], e1[Dn], e2[Dn];
    #pragma unroll
    for (int g = 0; g < Dn; ++g) ne[g] = node_emb[n * Dn + g];
    #pragma unroll
    for (int c = 0; c < Dn; ++c){
      float a1 = 0.f, a2 = 0.f;
      #pragma unroll
      for (int g = 0; g < Dn; ++g){ a1 += ne[g] * w1s[c][g]; a2 += ne[g] * w2s[c][g]; }
      e1[c] = a1 * INV_SQRT_TAU; e2[c] = a2 * INV_SQRT_TAU;
      NE1[(size_t)n * Dn + c] = e1[c];
      NE2[(size_t)n * Dn + c] = e2[c];
    }
    for (int m = 0; m < Mn; ++m){
      float dq = 0.f, dk = 0.f;
      #pragma unroll
      for (int c = 0; c < Dn; ++c){ dq += e1[c] * ps[m][c]; dk += e2[c] * ps[m][c]; }
      dq *= S4; dk *= S4;
      Dqn[(size_t)n * Mn + m] = dq;
      Dkn[(size_t)n * Mn + m] = dk;
      atomicMax(&bmax[m], fenc(dk));
    }
  }
  __syncthreads();
  if (tid < Mn) atomicMax(&maxkU[tid], bmax[tid]);
}

__global__ __launch_bounds__(32) void stabk_kernel(
    const unsigned int* __restrict__ maxkU, const float* __restrict__ Dbk, float* stabk)
{
  int b = blockIdx.x, lane = threadIdx.x;
  float v0 = fdec(maxkU[lane]) + Dbk[b * Mn + lane];
  float v1 = fdec(maxkU[lane + 32]) + Dbk[b * Mn + lane + 32];
  float v = wmaxr(fmaxf(v0, v1));
  if (lane == 0) stabk[b] = v;
}

// ---------------------------------------------------------------------------
// Input embedding: h0 = [input_w[:, :T]@x + markc | node_emb | te | we]
// Writes fp32 skip/h plus the bf16 mirror hbf used by the async GEMM staging.
__global__ __launch_bounds__(128) void embed_kernel(
    const float* __restrict__ x, const float* __restrict__ node_emb,
    const float* __restrict__ input_w, const float* __restrict__ markc,
    const float* __restrict__ te, const float* __restrict__ we,
    float* skip, float* h, unsigned short* hbf)
{
  __shared__ float Wsh[Dn][Tn];
  int b = blockIdx.y, n0 = blockIdx.x * 128, tid = threadIdx.x;
  for (int i = tid; i < Dn * Tn; i += 128){
    int c = i / Tn, t = i % Tn;
    Wsh[c][t] = input_w[c * (3 * Tn) + t];
  }
  __syncthreads();
  int n = n0 + tid;
  if (n >= Nn) return;
  float acc[Dn];
  #pragma unroll
  for (int c = 0; c < Dn; ++c) acc[c] = 0.f;
  for (int t = 0; t < Tn; ++t){
    float xv = x[((size_t)b * Tn + t) * Nn + n];
    #pragma unroll
    for (int c = 0; c < Dn; ++c) acc[c] += xv * Wsh[c][t];
  }
  size_t base = ((size_t)b * Nn + n) * Cn;
  #pragma unroll
  for (int c = 0; c < Dn; ++c){
    float v = acc[c] + markc[b * Dn + c];
    skip[base + c] = v; h[base + c] = v; hbf[base + c] = f2bf(v);
  }
  #pragma unroll
  for (int c = 0; c < Dn; ++c){
    float v = node_emb[(size_t)n * Dn + c];
    skip[base + Dn + c] = v; h[base + Dn + c] = v; hbf[base + Dn + c] = f2bf(v);
  }
  #pragma unroll
  for (int c = 0; c < Dn; ++c){
    float v = te[b * Dn + c];
    skip[base + 2 * Dn + c] = v; h[base + 2 * Dn + c] = v; hbf[base + 2 * Dn + c] = f2bf(v);
  }
  #pragma unroll
  for (int c = 0; c < Dn; ++c){
    float v = we[b * Dn + c];
    skip[base + 3 * Dn + c] = v; h[base + 3 * Dn + c] = v; hbf[base + 3 * Dn + c] = f2bf(v);
  }
}

// ---------------------------------------------------------------------------
// FAVOR+ features: one wave per (b,n).
// phi_q -> row-major bf16 [b][n][m]; phi_k -> tiled transposed bf16
// phikT[b][tile][m][row] (tile = n/128) so gate_kv can async-DMA it to LDS.
__global__ __launch_bounds__(256) void phi_kernel(
    const float* __restrict__ NE1, const float* __restrict__ NE2,
    const float* __restrict__ bias1, const float* __restrict__ bias2,
    const float* __restrict__ Dqn, const float* __restrict__ Dkn,
    const float* __restrict__ Dbq, const float* __restrict__ Dbk,
    const float* __restrict__ stabk,
    unsigned short* phiq16, unsigned short* phikT, float* ksum)
{
  int w = threadIdx.x >> 5, lane = threadIdx.x & 31;
  long g = (long)blockIdx.x * 8 + w;
  int b = (int)(g / Nn), n = (int)(g % Nn);
  float nv1 = NE1[(size_t)n * Dn + lane] + bias1[b * Dn + lane];
  float nv2 = NE2[(size_t)n * Dn + lane] + bias2[b * Dn + lane];
  float d1 = nv1 * S4, d2 = nv2 * S4;
  float diag_q = 0.5f * wsum(d1 * d1);
  float diag_k = 0.5f * wsum(d2 * d2);
  size_t nb = (size_t)b * Nn + n;

  float dq0 = Dqn[(size_t)n * Mn + lane]      + Dbq[b * Mn + lane];
  float dq1 = Dqn[(size_t)n * Mn + lane + 32] + Dbq[b * Mn + lane + 32];
  float rmax = wmaxr(fmaxf(dq0, dq1));
  float pq0 = RATIO * (__expf(dq0 - diag_q - rmax) + EPSF);   // exponent <= 0
  float pq1 = RATIO * (__expf(dq1 - diag_q - rmax) + EPSF);
  phiq16[nb * Mn + lane]      = f2bf(pq0);
  phiq16[nb * Mn + lane + 32] = f2bf(pq1);

  float st = stabk[b];
  float dk0 = Dkn[(size_t)n * Mn + lane]      + Dbk[b * Mn + lane];
  float dk1 = Dkn[(size_t)n * Mn + lane + 32] + Dbk[b * Mn + lane + 32];
  float pk0 = RATIO * (__expf(dk0 - diag_k - st) + EPSF);     // exponent <= 0
  float pk1 = RATIO * (__expf(dk1 - diag_k - st) + EPSF);
  int tile = n >> 7, row = n & 127;
  size_t tb = ((size_t)(b * NT + tile)) * Mn * 128;
  phikT[tb + (size_t)lane * 128 + row]        = f2bf(pk0);
  phikT[tb + (size_t)(lane + 32) * 128 + row] = f2bf(pk1);
  atomicAdd(&ksum[b * Mn + lane], pk0);
  atomicAdd(&ksum[b * Mn + lane + 32], pk1);
}

// ---------------------------------------------------------------------------
// Weight conversion: Wt[l][j][k] = bf16( j<128 ? in_w[l][j][k] : out_w[l][j-128][k] )
__global__ __launch_bounds__(256) void convw_kernel(
    const float* __restrict__ in_w, const float* __restrict__ out_w, unsigned short* Wt)
{
  int idx = blockIdx.x * 256 + threadIdx.x;
  if (idx >= Ln * 256 * Cn) return;
  int l = idx / (256 * Cn);
  int r = idx % (256 * Cn);
  int j = r / Cn, k = r % Cn;
  float v = (j < Cn) ? in_w[((size_t)l * Cn + j) * Cn + k]
                     : out_w[((size_t)l * Cn + (j - Cn)) * Cn + k];
  Wt[idx] = f2bf(v);
}

// ---------------------------------------------------------------------------
// Fused gating GEMM (WMMA bf16) + kv reduction (WMMA bf16, atomic accumulate).
// h tile and phi_k tile are DMA'd into LDS with global_load_async_to_lds_b128.
// u never leaves LDS.
__global__ __launch_bounds__(256) void gate_kv_kernel(
    const unsigned short* __restrict__ hbf, const unsigned short* __restrict__ phikT,
    const unsigned short* __restrict__ Wt, const float* __restrict__ in_b,
    const float* __restrict__ out_b, float* kv, int layer)
{
  __shared__ __align__(16) unsigned short hA[128][128];  // h tile bf16; later u^T
  __shared__ __align__(16) unsigned short pkT[64][128];  // phi_k transposed [m][row]
  int b = blockIdx.y, n0 = blockIdx.x * 128, tid = threadIdx.x;
  int nrows = min(128, Nn - n0);

  // Async DMA staging: 32KB h tile + 16KB phi_k tile, 16B per lane per op.
  {
    const unsigned short* gsrc = hbf + ((size_t)b * Nn + n0) * Cn;
    unsigned lbase = (unsigned)(size_t)(&hA[0][0]);
    #pragma unroll
    for (int it = 0; it < 8; ++it){
      int elem = (it * 256 + tid) * 8;                    // ushort index
      async_g2l_b128(lbase + (unsigned)(elem * 2), gsrc + elem);
    }
    const unsigned short* psrc = phikT + ((size_t)(b * NT + blockIdx.x)) * Mn * 128;
    unsigned pbase = (unsigned)(size_t)(&pkT[0][0]);
    #pragma unroll
    for (int it = 0; it < 4; ++it){
      int elem = (it * 256 + tid) * 8;
      async_g2l_b128(pbase + (unsigned)(elem * 2), psrc + elem);
    }
    wait_async0();
  }
  __syncthreads();

  int w = tid >> 5, lane = tid & 31, hi = lane >> 4, nlo = lane & 15;
  const unsigned short* Wl = Wt + (size_t)layer * 256 * Cn;
  float ureg[8][8];

  // Phase 1: wave w owns rows 16w..16w+15; 8 col tiles x 2 weights x 4 k-steps
  #pragma unroll
  for (int jt = 0; jt < 8; ++jt){
    int col = jt * 16 + nlo;
    v8f acc1 = splat8(in_b[layer * Cn + col]);
    v8f acc2 = splat8(out_b[layer * Cn + col]);
    #pragma unroll
    for (int ks = 0; ks < 4; ++ks){
      v16u a  = ldA(&hA[0][0], 128, 16 * w, 32 * ks);
      v16u b1 = ldB(Wl, Cn, jt * 16, 32 * ks);
      v16u b2 = ldB(Wl, Cn, Cn + jt * 16, 32 * ks);
      acc1 = wmma_bf16(a, b1, acc1);
      acc2 = wmma_bf16(a, b2, acc2);
    }
    #pragma unroll
    for (int r = 0; r < 8; ++r){
      float sg = 1.f / (1.f + __expf(-acc1[r]));          // branchless sigmoid
      ureg[jt][r] = sg * acc2[r];
    }
  }
  __syncthreads();
  // write u transposed into hA: hA[c][row_local]; pad rows forced to zero
  #pragma unroll
  for (int jt = 0; jt < 8; ++jt){
    #pragma unroll
    for (int r = 0; r < 8; ++r){
      int m = 16 * w + r + 8 * hi;
      int c = jt * 16 + nlo;
      hA[c][m] = (m < nrows) ? f2bf(ureg[jt][r]) : (unsigned short)0;
    }
  }
  __syncthreads();

  // Phase 2: kv[m,c] += sum_row phi_k[row,m] * u[row,c]
  int mt = w & 3, ch = w >> 2;
  #pragma unroll
  for (int ct = 0; ct < 4; ++ct){
    v8f acc = splat8(0.f);
    int cb = (4 * ch + ct) * 16;
    #pragma unroll
    for (int ks = 0; ks < 4; ++ks){
      v16u a  = ldA(&pkT[0][0], 128, 16 * mt, 32 * ks);
      v16u bb = ldB(&hA[0][0], 128, cb, 32 * ks);
      acc = wmma_bf16(a, bb, acc);
    }
    #pragma unroll
    for (int r = 0; r < 8; ++r){
      int m = 16 * mt + r + 8 * hi;
      int c = cb + nlo;
      atomicAdd(&kv[((size_t)b * Mn + m) * Cn + c], acc[r]);
    }
  }
}

// ---------------------------------------------------------------------------
// num = phi_q @ kv (WMMA), den = phi_q . ksum, residual add + LayerNorm,
// write fp32 h and bf16 mirror hbf.
__global__ __launch_bounds__(256) void attn_ln_kernel(
    const float* __restrict__ kv, const float* __restrict__ ksum,
    const unsigned short* __restrict__ phiq16, float* h, unsigned short* hbf,
    const float* __restrict__ ln_g, const float* __restrict__ ln_b, int layer)
{
  __shared__ __align__(16) unsigned short kvT[128][64];  // kv transposed [c][m] bf16
  __shared__ float denS[128];
  __shared__ float ksS[64];
  __shared__ float gS[128], bSs[128];
  int b = blockIdx.y, n0 = blockIdx.x * 128, tid = threadIdx.x;
  int nrows = min(128, Nn - n0);

  for (int i = tid; i < Mn * Cn; i += 256){
    int m = i >> 7, c = i & 127;
    kvT[c][m] = f2bf(kv[((size_t)b * Mn + m) * Cn + c]);
  }
  if (tid < 64) ksS[tid] = ksum[b * Mn + tid];
  if (tid < 128){ gS[tid] = ln_g[layer * Cn + tid]; bSs[tid] = ln_b[layer * Cn + tid]; }
  __syncthreads();

  int w = tid >> 5, lane = tid & 31, hi = lane >> 4, nlo = lane & 15;
  const unsigned short* phiBase = phiq16 + ((size_t)b * Nn + n0) * Mn;

  // den for this wave's 16 rows
  for (int r = 0; r < 16; ++r){
    int rl = 16 * w + r;
    float a = 0.f;
    if (rl < nrows){
      a = bf2f(phiBase[(size_t)rl * Mn + lane]) * ksS[lane]
        + bf2f(phiBase[(size_t)rl * Mn + lane + 32]) * ksS[lane + 32];
    }
    float s = wsum(a);
    if (lane == 0) denS[rl] = (rl < nrows && s != 0.f) ? s : 1.f;
  }
  __syncthreads();

  float uacc[8][8];
  #pragma unroll
  for (int jt = 0; jt < 8; ++jt){
    v8f acc = splat8(0.f);
    #pragma unroll
    for (int ks = 0; ks < 2; ++ks){
      v16u a  = ldA_guard(phiBase, Mn, 16 * w, 32 * ks, nrows);
      v16u bb = ldB(&kvT[0][0], Mn, jt * 16, 32 * ks);
      acc = wmma_bf16(a, bb, acc);
    }
    #pragma unroll
    for (int r = 0; r < 8; ++r) uacc[jt][r] = acc[r];
  }

  // residual + 1/den
  #pragma unroll
  for (int r = 0; r < 8; ++r){
    int rl = 16 * w + r + 8 * hi;
    float dinv = 1.f / denS[rl];
    bool valid = rl < nrows;
    #pragma unroll
    for (int jt = 0; jt < 8; ++jt){
      int c = jt * 16 + nlo;
      float res = valid ? h[((size_t)b * Nn + n0 + rl) * Cn + c] : 0.f;
      uacc[jt][r] = uacc[jt][r] * dinv + res;
    }
  }

  // LayerNorm per row (channels: 8 jt x 16 lanes within half) and store
  #pragma unroll
  for (int r = 0; r < 8; ++r){
    int rl = 16 * w + r + 8 * hi;
    float s = 0.f, q = 0.f;
    #pragma unroll
    for (int jt = 0; jt < 8; ++jt){ float v = uacc[jt][r]; s += v; q += v * v; }
    s = hsum16(s); q = hsum16(q);
    float mu = s * (1.f / 128.f);
    float var = q * (1.f / 128.f) - mu * mu;
    float rstd = rsqrtf(var + 1e-5f);
    if (rl < nrows){
      #pragma unroll
      for (int jt = 0; jt < 8; ++jt){
        int c = jt * 16 + nlo;
        float o = (uacc[jt][r] - mu) * rstd * gS[c] + bSs[c];
        size_t idx = ((size_t)b * Nn + n0 + rl) * Cn + c;
        h[idx] = o;
        hbf[idx] = f2bf(o);
      }
    }
  }
}

// ---------------------------------------------------------------------------
// Regression head: out[b,p,n] = [skip|h] . reg_w[p] + reg_b[p]
__global__ __launch_bounds__(128) void head_kernel(
    const float* __restrict__ skip, const float* __restrict__ h,
    const float* __restrict__ reg_w, const float* __restrict__ reg_b, float* out)
{
  __shared__ float rw[Pn][2 * Cn];
  __shared__ float rb[Pn];
  int b = blockIdx.y, n0 = blockIdx.x * 128, tid = threadIdx.x;
  for (int i = tid; i < Pn * 2 * Cn; i += 128) rw[i / (2 * Cn)][i % (2 * Cn)] = reg_w[i];
  if (tid < Pn) rb[tid] = reg_b[tid];
  __syncthreads();
  int n = n0 + tid;
  if (n >= Nn) return;
  size_t base = ((size_t)b * Nn + n) * Cn;
  float acc[Pn];
  #pragma unroll
  for (int p = 0; p < Pn; ++p) acc[p] = rb[p];
  for (int c = 0; c < Cn; ++c){
    float sv = skip[base + c];
    float hv = h[base + c];
    #pragma unroll
    for (int p = 0; p < Pn; ++p) acc[p] += sv * rw[p][c] + hv * rw[p][Cn + c];
  }
  #pragma unroll
  for (int p = 0; p < Pn; ++p)
    out[((size_t)b * Pn + p) * Nn + n] = acc[p];
}

// ---------------------------------------------------------------------------
extern "C" void kernel_launch(void* const* d_in, const int* in_sizes, int n_in,
                              void* d_out, int out_size, void* d_ws, size_t ws_size,
                              hipStream_t stream)
{
  (void)in_sizes; (void)n_in; (void)out_size; (void)ws_size;
  const float* x        = (const float*)d_in[0];
  const float* x_mark   = (const float*)d_in[1];
  const float* node_emb = (const float*)d_in[2];
  const float* time_tab = (const float*)d_in[3];
  const float* week_tab = (const float*)d_in[4];
  const float* input_w  = (const float*)d_in[5];
  const float* input_b  = (const float*)d_in[6];
  const float* w1_w     = (const float*)d_in[7];
  const float* w1_b     = (const float*)d_in[8];
  const float* w2_w     = (const float*)d_in[9];
  const float* w2_b     = (const float*)d_in[10];
  const float* in_w     = (const float*)d_in[11];
  const float* in_b     = (const float*)d_in[12];
  const float* out_w    = (const float*)d_in[13];
  const float* out_b    = (const float*)d_in[14];
  const float* ln_g     = (const float*)d_in[15];
  const float* ln_b     = (const float*)d_in[16];
  const float* reg_w    = (const float*)d_in[17];
  const float* reg_b    = (const float*)d_in[18];
  const float* proj     = (const float*)d_in[19];
  float* out = (float*)d_out;

  char* ws = (char*)d_ws;
  size_t off = 0;
  auto alloc = [&](size_t bytes) -> char* {
    char* p = ws + off;
    off = (off + bytes + 255) & ~(size_t)255;
    return p;
  };
  float*          skip   = (float*)alloc((size_t)Bn * Nn * Cn * 4);
  float*          hbuf   = (float*)alloc((size_t)Bn * Nn * Cn * 4);
  unsigned short* hbf    = (unsigned short*)alloc(((size_t)Bn * Nn + 128) * Cn * 2); // +slack for tail tile DMA
  unsigned short* phiq16 = (unsigned short*)alloc((size_t)Bn * Nn * Mn * 2);
  unsigned short* phikT  = (unsigned short*)alloc((size_t)Bn * NT * Mn * 128 * 2);
  float*          NE1    = (float*)alloc((size_t)Nn * Dn * 4);
  float*          NE2    = (float*)alloc((size_t)Nn * Dn * 4);
  float*          Dqn    = (float*)alloc((size_t)Nn * Mn * 4);
  float*          Dkn    = (float*)alloc((size_t)Nn * Mn * 4);
  unsigned short* Wt     = (unsigned short*)alloc((size_t)Ln * 256 * Cn * 2);
  float*          kv     = (float*)alloc((size_t)Bn * Mn * Cn * 4);
  float*          ksum   = (float*)alloc((size_t)Bn * Mn * 4);
  float*          te     = (float*)alloc((size_t)Bn * Dn * 4);
  float*          we     = (float*)alloc((size_t)Bn * Dn * 4);
  float*          markc  = (float*)alloc((size_t)Bn * Dn * 4);
  float*          bias1  = (float*)alloc((size_t)Bn * Dn * 4);
  float*          bias2  = (float*)alloc((size_t)Bn * Dn * 4);
  float*          Dbq    = (float*)alloc((size_t)Bn * Mn * 4);
  float*          Dbk    = (float*)alloc((size_t)Bn * Mn * 4);
  unsigned int*   maxkU  = (unsigned int*)alloc((size_t)Mn * 4);
  float*          stabk  = (float*)alloc((size_t)Bn * 4);

  hipMemsetAsync(maxkU, 0, (size_t)Mn * 4, stream);
  hipMemsetAsync(ksum, 0, (size_t)Bn * Mn * 4, stream);
  hipMemsetAsync(phikT, 0, (size_t)Bn * NT * Mn * 128 * 2, stream);   // zero-pads tail rows

  prep_batch_kernel<<<Bn, 64, 0, stream>>>(x_mark, time_tab, week_tab, input_w,
      input_b, w1_w, w1_b, w2_w, w2_b, proj, te, we, markc, bias1, bias2, Dbq, Dbk);
  prep_node_kernel<<<(Nn + 63) / 64, 64, 0, stream>>>(node_emb, w1_w, w2_w, proj,
      NE1, NE2, Dqn, Dkn, maxkU);
  stabk_kernel<<<Bn, 32, 0, stream>>>(maxkU, Dbk, stabk);

  dim3 gE(NT, Bn);
  embed_kernel<<<gE, 128, 0, stream>>>(x, node_emb, input_w, markc, te, we,
                                       skip, hbuf, hbf);
  phi_kernel<<<(Bn * Nn) / 8, 256, 0, stream>>>(NE1, NE2, bias1, bias2, Dqn, Dkn,
      Dbq, Dbk, stabk, phiq16, phikT, ksum);
  convw_kernel<<<(Ln * 256 * Cn + 255) / 256, 256, 0, stream>>>(in_w, out_w, Wt);

  for (int l = 0; l < Ln; ++l){
    hipMemsetAsync(kv, 0, (size_t)Bn * Mn * Cn * 4, stream);
    gate_kv_kernel<<<gE, 256, 0, stream>>>(hbf, phikT, Wt, in_b, out_b, kv, l);
    attn_ln_kernel<<<gE, 256, 0, stream>>>(kv, ksum, phiq16, hbuf, hbf, ln_g, ln_b, l);
  }
  head_kernel<<<gE, 128, 0, stream>>>(skip, hbuf, reg_w, reg_b, out);
}
